// AttnMixer_45956150067710
// MI455X (gfx1250) — compile-verified, matrix operand order checked by reading, not run on hardware
//
#include <hip/hip_runtime.h>
#include <hip/hip_bf16.h>

typedef __attribute__((ext_vector_type(16))) _Float16 v16h;
typedef __attribute__((ext_vector_type(8)))  _Float16 v8h;
typedef __attribute__((ext_vector_type(4)))  _Float16 v4h;
typedef __attribute__((ext_vector_type(8)))  float    v8f;
typedef __attribute__((ext_vector_type(4)))  int      v4i;

// ---- constants of the problem ----
#define NB       32          // batch
#define HH       16          // image rows
#define WW       64          // image cols
#define SS       1024        // H*W
#define CC       512         // embed dims
#define NHEAD    8
#define HD       64          // head dim
#define SCALE    0.125f      // HD^-0.5

// ---- gfx1250 async global->LDS copy (guarded; falls back to sync staging) ----
#if defined(__AMDGCN__) && __has_builtin(__builtin_amdgcn_global_load_async_to_lds_b128)
#define HAVE_ASYNC 1
#else
#define HAVE_ASYNC 0
#endif

#if HAVE_ASYNC
typedef __attribute__((address_space(1))) v4i* as1_v4i_p;   // global (prints as "__device__")
typedef __attribute__((address_space(3))) v4i* as3_v4i_p;   // LDS    (prints as "__shared__")
#endif

__device__ __forceinline__ void stage16(const _Float16* g, _Float16* l) {
#if HAVE_ASYNC
    __builtin_amdgcn_global_load_async_to_lds_b128(
        (as1_v4i_p)(void*)g, (as3_v4i_p)(void*)l, 0, 0);
#else
    *(v8h*)l = *(const v8h*)g;
#endif
}

__device__ __forceinline__ void wait_async() {
#if HAVE_ASYNC
#if __has_builtin(__builtin_amdgcn_s_wait_asynccnt)
    __builtin_amdgcn_s_wait_asynccnt(0);
#else
    asm volatile("s_wait_asynccnt 0x0" ::: "memory");
#endif
#endif
}

__device__ __forceinline__ v16h ld_frag(const _Float16* p0, const _Float16* p1) {
    v8h lo = *(const v8h*)p0;
    v8h hi = *(const v8h*)p1;
    v16h a;
#pragma unroll
    for (int i = 0; i < 8; ++i) { a[i] = lo[i]; a[i + 8] = hi[i]; }
    return a;
}

__device__ __forceinline__ v8f wmma16(v16h a, v16h b, v8f c) {
    return __builtin_amdgcn_wmma_f32_16x16x32_f16(false, a, false, b, (short)0, c, false, false);
}

// ------------------- conversion kernels -------------------
__global__ void cvt_x_f32_to_f16(const float* __restrict__ x, _Float16* __restrict__ xh, int n4) {
    int i = blockIdx.x * blockDim.x + threadIdx.x;
    if (i < n4) {
        float4 v = ((const float4*)x)[i];
        v4h h; h[0] = (_Float16)v.x; h[1] = (_Float16)v.y; h[2] = (_Float16)v.z; h[3] = (_Float16)v.w;
        ((v4h*)xh)[i] = h;
    }
}

// w: [K][N] row-major f32  ->  wt: [N][K] f16
__global__ void cvt_w_transpose(const float* __restrict__ w, _Float16* __restrict__ wt, int K, int N) {
    int i = blockIdx.x * blockDim.x + threadIdx.x;
    if (i < K * N) {
        int c = i / K;
        int k = i - c * K;
        wt[i] = (_Float16)w[k * N + c];
    }
}

// ------------------- QKV GEMM: Xh[32768][512] @ WqkvT -> scatter Q,K,Vt -------------------
// 4-wave workgroup computes a 64x64 tile; B panel (64 cols x 32 k) staged in LDS,
// double-buffered via async global->LDS copies. grid = 512 (m64) * 24 (n64) blocks.
__global__ void __launch_bounds__(128) gemm_qkv(const _Float16* __restrict__ Xh,
                                                const _Float16* __restrict__ Wt,   // [1536][512]
                                                _Float16* __restrict__ Qs,
                                                _Float16* __restrict__ Kf,
                                                _Float16* __restrict__ Vt) {
    __shared__ __align__(16) _Float16 Bp[2 * 64 * 32];   // double-buffered B panel, 8 KB

    const int tid = threadIdx.x;
    const int wave = tid >> 5;
    const int l = tid & 31;
    const int nlane = l & 15;
    const int g = (l >> 4) & 1;
    const int b = blockIdx.x;
    const int mb = b & 511;        // 512 m-blocks of 64 rows
    const int nt = b >> 9;         // 24 n-blocks of 64 cols
    const int ntbase = nt * 64;

    const _Float16* arow = Xh + ((size_t)(mb * 64 + wave * 16 + nlane)) * CC;

    const v8f vzero = {0.f, 0.f, 0.f, 0.f, 0.f, 0.f, 0.f, 0.f};
    v8f acc[4];
#pragma unroll
    for (int ct = 0; ct < 4; ++ct) acc[ct] = vzero;

    // stage chunk 0
#pragma unroll
    for (int i = 0; i < 2; ++i) {
        int e = i * 128 + tid;            // 0..255
        int r = e >> 2, seg = e & 3;
        stage16(Wt + ((size_t)(ntbase + r)) * CC + seg * 8, &Bp[r * 32 + seg * 8]);
    }
    wait_async();
    __syncthreads();

#pragma unroll 2
    for (int kc = 0; kc < 16; ++kc) {
        const int cur = kc & 1;
        if (kc < 15) {
            _Float16* nbuf = &Bp[(cur ^ 1) * 2048];
#pragma unroll
            for (int i = 0; i < 2; ++i) {
                int e = i * 128 + tid;
                int r = e >> 2, seg = e & 3;
                stage16(Wt + ((size_t)(ntbase + r)) * CC + (kc + 1) * 32 + seg * 8,
                        &nbuf[r * 32 + seg * 8]);
            }
        }
        if (kc < 15) __builtin_prefetch(arow + (kc + 1) * 32, 0, 3);

        v16h a = ld_frag(arow + kc * 32 + g * 8, arow + kc * 32 + g * 8 + 16);
        const _Float16* bpan = &Bp[cur * 2048];
#pragma unroll
        for (int ct = 0; ct < 4; ++ct) {
            const _Float16* brow = bpan + (ct * 16 + nlane) * 32;
            v16h bb = ld_frag(brow + g * 8, brow + g * 8 + 16);
            acc[ct] = wmma16(a, bb, acc[ct]);
        }
        wait_async();
        __syncthreads();
    }

    // scatter epilogue: col c in [0,1536): t=c>>9 (0:q 1:k 2:v), head=(c>>6)&7, d=c&63
#pragma unroll
    for (int ct = 0; ct < 4; ++ct) {
#pragma unroll
        for (int j = 0; j < 8; ++j) {
            int m = j + 8 * g;
            size_t row = (size_t)mb * 64 + wave * 16 + m;   // 0..32767
            int nbatch = (int)(row >> 10);
            int s = (int)(row & 1023);
            int c = ntbase + ct * 16 + nlane;               // 0..1535
            int t = c >> 9;
            int head = (c >> 6) & 7;
            int d = c & 63;
            float v = acc[ct][j];
            if (t == 0) {
                Qs[(((size_t)nbatch * NHEAD + head) * SS + s) * HD + d] = (_Float16)(v * SCALE);
            } else if (t == 1) {
                Kf[(((size_t)nbatch * NHEAD + head) * SS + s) * HD + d] = (_Float16)v;
            } else {
                Vt[(((size_t)nbatch * NHEAD + head) * HD + d) * SS + s] = (_Float16)v;
            }
        }
    }
}

// ------------------- local attention: one wave per (n, head, qtile16) -------------------
__global__ void __launch_bounds__(32) attn_local(const _Float16* __restrict__ Qs,
                                                 const _Float16* __restrict__ Kf,
                                                 const _Float16* __restrict__ Vt,
                                                 _Float16* __restrict__ AO) {
    __shared__ __align__(16) _Float16 Plds[16 * 32];   // P tile: 16 rows x 32 keys

    const int l = threadIdx.x;
    const int nlane = l & 15;
    const int g = l >> 4;
    const int b = blockIdx.x;
    const int qt = b & 63;
    const int head = (b >> 6) & 7;
    const int nbatch = b >> 9;

    const int qh = qt >> 2;
    const int qw0 = (qt & 3) << 4;

    const _Float16* Qbase = Qs + (((size_t)nbatch * NHEAD + head) * SS + qt * 16) * HD;
    const _Float16* Kbase = Kf + ((size_t)nbatch * NHEAD + head) * SS * HD;
    const _Float16* Vbase = Vt + ((size_t)nbatch * NHEAD + head) * HD * SS;

    // Q fragments (A layout): lane holds row M=nlane; K chunks 0..31 and 32..63
    const _Float16* qrow = Qbase + (size_t)nlane * HD;
    v16h a0 = ld_frag(qrow + g * 8,      qrow + g * 8 + 16);
    v16h a1 = ld_frag(qrow + g * 8 + 32, qrow + g * 8 + 48);

    const v8f vzero = {0.f, 0.f, 0.f, 0.f, 0.f, 0.f, 0.f, 0.f};
    v8f acc[4];
#pragma unroll
    for (int ct = 0; ct < 4; ++ct) acc[ct] = vzero;
    float vsum[8];
#pragma unroll
    for (int j = 0; j < 8; ++j) vsum[j] = 0.f;

    int khs = qh - 3; if (khs < 0) khs = 0;
    int khe = qh + 3; if (khe > 15) khe = 15;
    int wts = (qw0 - 5) < 0 ? 0 : ((qw0 - 5) >> 4);
    int wte = (qw0 + 20) >> 4; if (wte > 3) wte = 3;
    const int nwt = wte - wts + 1;
    const int T = (khe - khs + 1) * nwt;

    for (int p = 0; p < T; p += 2) {
        int s0[2], wtv[2];
        const int valid1 = (p + 1) < T;
#pragma unroll
        for (int u = 0; u < 2; ++u) {
            int t = p + u; if (t >= T) t = p;
            int kh = khs + t / nwt;
            int wt = wts + t % nwt;
            s0[u] = kh * 64 + wt * 16;
            wtv[u] = wt;
        }

        // ---- score tiles (QK^T), mask, exp, stage P into LDS ----
#pragma unroll
        for (int u = 0; u < 2; ++u) {
            if (u == 1 && !valid1) {
#pragma unroll
                for (int j = 0; j < 8; ++j) {
                    int m = j + 8 * g;
                    Plds[m * 32 + 16 + nlane] = (_Float16)0.0f;
                }
                continue;
            }
            const _Float16* krow = Kbase + (size_t)(s0[u] + nlane) * HD;  // B lane holds key col N=nlane
            v16h b0 = ld_frag(krow + g * 8,      krow + g * 8 + 16);
            v16h b1 = ld_frag(krow + g * 8 + 32, krow + g * 8 + 48);
            v8f c = vzero;
            c = wmma16(a0, b0, c);
            c = wmma16(a1, b1, c);
            int kw = wtv[u] * 16 + nlane;
#pragma unroll
            for (int j = 0; j < 8; ++j) {
                int m = j + 8 * g;
                int dw = kw - (qw0 + m); if (dw < 0) dw = -dw;
                float pv = (dw <= 5) ? __expf(c[j]) : 0.0f;
                vsum[j] += pv;
                Plds[m * 32 + u * 16 + nlane] = (_Float16)pv;
            }
        }
        __syncthreads();

        // ---- P (16x32 from LDS, A layout) @ V (32x64 from Vt) ----
        const _Float16* prow = &Plds[nlane * 32];
        v16h ap = ld_frag(prow + g * 8, prow + g * 8 + 16);
#pragma unroll
        for (int ct = 0; ct < 4; ++ct) {
            const _Float16* vrow = Vbase + (size_t)(ct * 16 + nlane) * SS;
            // halves 0..7 -> K=g*8+{0..7} (tile 0), halves 8..15 -> K=16+g*8+{0..7} (tile 1)
            v16h bv = ld_frag(vrow + s0[0] + g * 8, vrow + s0[1] + g * 8);
            acc[ct] = wmma16(ap, bv, acc[ct]);
        }
        __syncthreads();
    }

    // row sums: reduce across the 16 lanes holding the same M (xor within half-groups)
#pragma unroll
    for (int j = 0; j < 8; ++j) {
        float v = vsum[j];
        v += __shfl_xor(v, 1, 32);
        v += __shfl_xor(v, 2, 32);
        v += __shfl_xor(v, 4, 32);
        v += __shfl_xor(v, 8, 32);
        vsum[j] = v;
    }

    // normalize + store to AO[32768][512] f16 (head-concatenated layout)
    size_t rowbase = (size_t)nbatch * SS + qt * 16;
#pragma unroll
    for (int j = 0; j < 8; ++j) {
        float rinv = 1.0f / vsum[j];
        int m = j + 8 * g;
        size_t row = rowbase + m;
#pragma unroll
        for (int ct = 0; ct < 4; ++ct) {
            AO[row * CC + head * HD + ct * 16 + nlane] = (_Float16)(acc[ct][j] * rinv);
        }
    }
}

// ------------------- projection GEMM: AO[32768][512] @ WprojT + bias -> f32 out -------------------
// same 4-wave / LDS-staged structure as gemm_qkv. grid = 512 (m64) * 8 (n64).
__global__ void __launch_bounds__(128) gemm_proj(const _Float16* __restrict__ AO,
                                                 const _Float16* __restrict__ Wt,   // [512][512]
                                                 const float* __restrict__ bias,
                                                 float* __restrict__ out) {
    __shared__ __align__(16) _Float16 Bp[2 * 64 * 32];

    const int tid = threadIdx.x;
    const int wave = tid >> 5;
    const int l = tid & 31;
    const int nlane = l & 15;
    const int g = (l >> 4) & 1;
    const int b = blockIdx.x;
    const int mb = b & 511;
    const int nt = b >> 9;          // 0..7
    const int ntbase = nt * 64;

    const _Float16* arow = AO + ((size_t)(mb * 64 + wave * 16 + nlane)) * CC;

    const v8f vzero = {0.f, 0.f, 0.f, 0.f, 0.f, 0.f, 0.f, 0.f};
    v8f acc[4];
#pragma unroll
    for (int ct = 0; ct < 4; ++ct) acc[ct] = vzero;

#pragma unroll
    for (int i = 0; i < 2; ++i) {
        int e = i * 128 + tid;
        int r = e >> 2, seg = e & 3;
        stage16(Wt + ((size_t)(ntbase + r)) * CC + seg * 8, &Bp[r * 32 + seg * 8]);
    }
    wait_async();
    __syncthreads();

#pragma unroll 2
    for (int kc = 0; kc < 16; ++kc) {
        const int cur = kc & 1;
        if (kc < 15) {
            _Float16* nbuf = &Bp[(cur ^ 1) * 2048];
#pragma unroll
            for (int i = 0; i < 2; ++i) {
                int e = i * 128 + tid;
                int r = e >> 2, seg = e & 3;
                stage16(Wt + ((size_t)(ntbase + r)) * CC + (kc + 1) * 32 + seg * 8,
                        &nbuf[r * 32 + seg * 8]);
            }
        }
        if (kc < 15) __builtin_prefetch(arow + (kc + 1) * 32, 0, 3);

        v16h a = ld_frag(arow + kc * 32 + g * 8, arow + kc * 32 + g * 8 + 16);
        const _Float16* bpan = &Bp[cur * 2048];
#pragma unroll
        for (int ct = 0; ct < 4; ++ct) {
            const _Float16* brow = bpan + (ct * 16 + nlane) * 32;
            v16h bb = ld_frag(brow + g * 8, brow + g * 8 + 16);
            acc[ct] = wmma16(a, bb, acc[ct]);
        }
        wait_async();
        __syncthreads();
    }

#pragma unroll
    for (int ct = 0; ct < 4; ++ct) {
#pragma unroll
        for (int j = 0; j < 8; ++j) {
            int m = j + 8 * g;
            size_t row = (size_t)mb * 64 + wave * 16 + m;
            int col = ntbase + ct * 16 + nlane;
            out[row * CC + col] = acc[ct][j] + bias[col];
        }
    }
}

// ------------------- host-side launch -------------------
extern "C" void kernel_launch(void* const* d_in, const int* in_sizes, int n_in,
                              void* d_out, int out_size, void* d_ws, size_t ws_size,
                              hipStream_t stream) {
    const float* x     = (const float*)d_in[0];   // [32,1024,512]
    const float* Wqkv  = (const float*)d_in[1];   // [512,1536]
    const float* Wproj = (const float*)d_in[2];   // [512,512]
    const float* bproj = (const float*)d_in[3];   // [512]
    float* out = (float*)d_out;

    // workspace layout (f16 elements)
    _Float16* Xh  = (_Float16*)d_ws;              // 32768*512   = 16,777,216
    _Float16* Wqt = Xh  + (size_t)32768 * 512;    // 1536*512    =    786,432
    _Float16* Wpt = Wqt + (size_t)1536 * 512;     // 512*512     =    262,144
    _Float16* Qs  = Wpt + (size_t)512 * 512;      // 32*8*1024*64
    _Float16* Kf  = Qs  + (size_t)NB * NHEAD * SS * HD;
    _Float16* Vt  = Kf  + (size_t)NB * NHEAD * SS * HD;
    _Float16* AO  = Vt  + (size_t)NB * NHEAD * SS * HD;   // 32768*512

    // 1) convert inputs
    {
        int n4 = (32768 * 512) / 4;
        cvt_x_f32_to_f16<<<(n4 + 255) / 256, 256, 0, stream>>>(x, Xh, n4);
        int nq = 512 * 1536;
        cvt_w_transpose<<<(nq + 255) / 256, 256, 0, stream>>>(Wqkv, Wqt, 512, 1536);
        int np = 512 * 512;
        cvt_w_transpose<<<(np + 255) / 256, 256, 0, stream>>>(Wproj, Wpt, 512, 512);
    }

    // 2) QKV GEMM + scatter (512 m64-blocks x 24 n64-blocks, 4 waves each)
    gemm_qkv<<<512 * 24, 128, 0, stream>>>(Xh, Wqt, Qs, Kf, Vt);

    // 3) local attention (32 batch * 8 heads * 64 q-tiles)
    attn_local<<<NB * NHEAD * (SS / 16), 32, 0, stream>>>(Qs, Kf, Vt, AO);

    // 4) projection GEMM (512 m64-blocks x 8 n64-blocks, 4 waves each)
    gemm_proj<<<512 * 8, 128, 0, stream>>>(AO, Wpt, bproj, out);
}
